// LinearAttention_33097017983568
// MI455X (gfx1250) — matrix operand order repaired and checked by reading, not compiled, not run on hardware
//
#include <hip/hip_runtime.h>

constexpr int kCH  = 128;
constexpr int kMID = 32;
constexpr int kB   = 16;
constexpr int kN   = 16384;   // H*W
constexpr int kNT  = 128;     // n-tile per workgroup

typedef __attribute__((ext_vector_type(16))) _Float16 v16h;
typedef __attribute__((ext_vector_type(8)))  _Float16 v8h;
typedef __attribute__((ext_vector_type(8)))  float    v8f;

// 16 contiguous halves -> v16h (two 16B DS loads)
__device__ inline v16h ld16c(const _Float16* p) {
  v8h lo = *(const v8h*)p;
  v8h hi = *(const v8h*)(p + 8);
  v16h r;
#pragma unroll
  for (int i = 0; i < 8; ++i) { r[i] = lo[i]; r[i + 8] = hi[i]; }
  return r;
}
// two groups of 8 halves (A-operand K groups {base, base+16})
__device__ inline v16h ld16g(const _Float16* p0, const _Float16* p1) {
  v8h lo = *(const v8h*)p0;
  v8h hi = *(const v8h*)p1;
  v16h r;
#pragma unroll
  for (int i = 0; i < 8; ++i) { r[i] = lo[i]; r[i + 8] = hi[i]; }
  return r;
}
__device__ inline v8f wmma16(v16h a, v16h b, v8f c) {
  // D = A(16x32 f16) * B(32x16 f16) + C(f32)
  return __builtin_amdgcn_wmma_f32_16x16x32_f16(false, a, false, b, (short)0,
                                                c, false, false);
}

// ---------------------------------------------------------------------------
// Kernel 1: Q/K/V projections (WMMA f16), phi, Q->ws, per-tile KV & Ksum
// grid = (kN/kNT, kB), block = 256 (8 waves)
// ---------------------------------------------------------------------------
__global__ __launch_bounds__(256) void k_qkv(
    const float* __restrict__ x, const float* __restrict__ Wq,
    const float* __restrict__ Wk, const float* __restrict__ Wv,
    float* __restrict__ gQ, float* __restrict__ gKV, float* __restrict__ gKs)
{
  __shared__ _Float16 xs[kNT][kCH + 8];        // x tile, transposed [n][c]
  __shared__ _Float16 wsm[3][kMID][kCH + 8];   // Wq/Wk/Wv as f16 [m][c]
  __shared__ _Float16 kf[kMID][kNT + 8];       // phi(K) tile [m][n]
  __shared__ _Float16 vf[kMID][kNT + 8];       // V tile [m][n]
  __shared__ float    ksum[kMID];

  const int tid  = threadIdx.x;
  const int wave = tid >> 5, lane = tid & 31;
  const int l16  = lane & 15;
  const bool hi  = lane >= 16;
  const int b    = blockIdx.y;
  const int n0   = blockIdx.x * kNT;

  // stage weights (f32 -> f16)
  for (int i = tid; i < 3 * kMID * kCH; i += 256) {
    int mat = i / (kMID * kCH);
    int r   = i - mat * (kMID * kCH);
    int m   = r / kCH, c = r - m * kCH;
    const float* W = (mat == 0) ? Wq : (mat == 1) ? Wk : Wv;
    wsm[mat][m][c] = (_Float16)W[m * kCH + c];
  }
  if (tid < kMID) ksum[tid] = 0.f;

  // stage x tile transposed [n][c] (coalesced reads along n)
  const float* xb = x + (size_t)b * kCH * kN;
  for (int i = tid; i < kCH * kNT; i += 256) {
    int c = i / kNT, n = i - c * kNT;
    int j = i + 2048;
    if (j < kCH * kNT) {  // software prefetch ahead (global_prefetch_b8)
      int c2 = j / kNT, n2 = j - c2 * kNT;
      __builtin_prefetch(&xb[(size_t)c2 * kN + n0 + n2], 0, 0);
    }
    xs[n][c] = (_Float16)xb[(size_t)c * kN + n0 + n];
  }
  __syncthreads();

  // each wave owns a 16-wide n slab
  const int nb = wave * 16;
  float krow[2][8];
#pragma unroll
  for (int mt = 0; mt < 2; ++mt)
#pragma unroll
    for (int r = 0; r < 8; ++r) krow[mt][r] = 0.f;

  for (int mat = 0; mat < 3; ++mat) {
#pragma unroll
    for (int mt = 0; mt < 2; ++mt) {
      v8f acc;
#pragma unroll
      for (int r = 0; r < 8; ++r) acc[r] = 0.f;
#pragma unroll
      for (int kc = 0; kc < kCH / 32; ++kc) {
        const _Float16* ap = &wsm[mat][mt * 16 + l16][kc * 32 + (hi ? 8 : 0)];
        v16h a = ld16g(ap, ap + 16);
        const _Float16* bp = &xs[nb + l16][kc * 32 + (hi ? 8 : 0)];
        v16h bv = ld16g(bp, bp + 16);
        acc = wmma16(a, bv, acc);
      }
      if (mat < 2) {  // phi = elu(x)+1 : x>0 ? x+1 : exp(x)
#pragma unroll
        for (int r = 0; r < 8; ++r) {
          float t = acc[r];
          acc[r]  = (t > 0.f) ? (t + 1.f) : __expf(t);
        }
      }
      const int m0   = mt * 16 + (hi ? 8 : 0);
      const int ncol = nb + l16;
      if (mat == 0) {
#pragma unroll
        for (int r = 0; r < 8; ++r)
          gQ[((size_t)b * kMID + m0 + r) * kN + n0 + ncol] = acc[r];
      } else if (mat == 1) {
#pragma unroll
        for (int r = 0; r < 8; ++r) {
          kf[m0 + r][ncol] = (_Float16)acc[r];
          krow[mt][r] += acc[r];
        }
      } else {
#pragma unroll
        for (int r = 0; r < 8; ++r) vf[m0 + r][ncol] = (_Float16)acc[r];
      }
    }
  }

  // K row-sum: reduce across the 16 lanes sharing a row, then LDS atomic
#pragma unroll
  for (int mt = 0; mt < 2; ++mt)
#pragma unroll
    for (int r = 0; r < 8; ++r) {
      float v = krow[mt][r];
      v += __shfl_xor(v, 1, 16);
      v += __shfl_xor(v, 2, 16);
      v += __shfl_xor(v, 4, 16);
      v += __shfl_xor(v, 8, 16);
      if (l16 == 0) atomicAdd(&ksum[mt * 16 + (hi ? 8 : 0) + r], v);
    }
  __syncthreads();

  // KV += K * V^T over this tile's n range (WMMA, f32 accumulation)
  {
    const int tile = wave & 3, half = wave >> 2;
    const int mi = tile >> 1, vi = tile & 1;
    v8f acc;
#pragma unroll
    for (int r = 0; r < 8; ++r) acc[r] = 0.f;
    const int c0 = half * (kNT / 64), c1 = c0 + kNT / 64;
    for (int nc = c0; nc < c1; ++nc) {
      const _Float16* ap = &kf[mi * 16 + l16][nc * 32 + (hi ? 8 : 0)];
      v16h a = ld16g(ap, ap + 16);
      const _Float16* bp = &vf[vi * 16 + l16][nc * 32 + (hi ? 8 : 0)];
      v16h bv = ld16g(bp, bp + 16);
      acc = wmma16(a, bv, acc);
    }
#pragma unroll
    for (int r = 0; r < 8; ++r) {
      int m = mi * 16 + (hi ? 8 : 0) + r;
      int v = vi * 16 + l16;
      atomicAdd(&gKV[((size_t)b * kMID + m) * kMID + v], acc[r]);
    }
  }
  __syncthreads();
  if (tid < kMID) atomicAdd(&gKs[b * kMID + tid], ksum[tid]);
}

// ---------------------------------------------------------------------------
// Kernel 2: out = (KV@Q)/(n+eps)/denom ; y = Wout@out ; BN partial sums
// grid = (kN/kNT, kB), block = 256
// ---------------------------------------------------------------------------
__global__ __launch_bounds__(256) void k_attn(
    const float* __restrict__ gQ, const float* __restrict__ gKV,
    const float* __restrict__ gKs, const float* __restrict__ Wout,
    float* __restrict__ y, float* __restrict__ gS1, float* __restrict__ gS2)
{
  __shared__ _Float16 kvh[kMID][kMID + 8];   // A of GEMM1  [m][v]
  __shared__ _Float16 qT[kNT][kMID + 8];     // B of GEMM1  [n][v]
  __shared__ _Float16 outT[kNT][kMID + 8];   // B of GEMM2  [n][m]
  __shared__ _Float16 wo[kCH][kMID + 8];     // A of GEMM2  [c][m]
  __shared__ float    den[kNT];
  __shared__ float    ksl[kMID];
  __shared__ float    s1[kCH], s2[kCH];

  const int tid  = threadIdx.x;
  const int wave = tid >> 5, lane = tid & 31;
  const int l16  = lane & 15;
  const bool hi  = lane >= 16;
  const int b    = blockIdx.y;
  const int n0   = blockIdx.x * kNT;

  if (tid < kMID) ksl[tid] = gKs[b * kMID + tid];
  for (int i = tid; i < kMID * kMID; i += 256)
    kvh[i / kMID][i % kMID] = (_Float16)gKV[(size_t)b * kMID * kMID + i];
  for (int i = tid; i < kCH * kMID; i += 256)
    wo[i / kMID][i % kMID] = (_Float16)Wout[i];
  if (tid < kNT) den[tid] = 1e-6f;
  if (tid < kCH) { s1[tid] = 0.f; s2[tid] = 0.f; }
  __syncthreads();

  // Q tile: transpose into LDS + accumulate denom[n] = sum_m Q[m,n]*Ks[m]
  for (int i = tid; i < kMID * kNT; i += 256) {
    int m = i / kNT, n = i - m * kNT;
    float q = gQ[((size_t)b * kMID + m) * kN + n0 + n];
    qT[n][m] = (_Float16)q;
    atomicAdd(&den[n], q * ksl[m]);
  }
  __syncthreads();

  const float inv_n = 1.f / ((float)kN + 1e-6f);
  // GEMM1: out[m][n] = KV @ Q   (K = 32 -> single WMMA per tile)
#pragma unroll
  for (int mt = 0; mt < 2; ++mt) {
    const _Float16* ap = &kvh[mt * 16 + l16][hi ? 8 : 0];
    v16h a = ld16g(ap, ap + 16);
    const _Float16* bp = &qT[wave * 16 + l16][hi ? 8 : 0];
    v16h bv = ld16g(bp, bp + 16);
    v8f acc;
#pragma unroll
    for (int r = 0; r < 8; ++r) acc[r] = 0.f;
    acc = wmma16(a, bv, acc);
    const float sc = inv_n / den[wave * 16 + l16];
    const int m0 = mt * 16 + (hi ? 8 : 0);
#pragma unroll
    for (int r = 0; r < 8; ++r)
      outT[wave * 16 + l16][m0 + r] = (_Float16)(acc[r] * sc);
  }
  __syncthreads();

  // GEMM2: y[c][n] = Wout @ out ; write y ; BN partial sums
  for (int ct = 0; ct < kCH / 16; ++ct) {
    const _Float16* ap = &wo[ct * 16 + l16][hi ? 8 : 0];
    v16h a = ld16g(ap, ap + 16);
    const _Float16* bp = &outT[wave * 16 + l16][hi ? 8 : 0];
    v16h bv = ld16g(bp, bp + 16);
    v8f acc;
#pragma unroll
    for (int r = 0; r < 8; ++r) acc[r] = 0.f;
    acc = wmma16(a, bv, acc);
    const int c0 = ct * 16 + (hi ? 8 : 0);
    const int n  = n0 + wave * 16 + l16;
#pragma unroll
    for (int r = 0; r < 8; ++r) {
      float v = acc[r];
      y[((size_t)b * kCH + c0 + r) * kN + n] = v;
      float a1 = v, a2 = v * v;
      a1 += __shfl_xor(a1, 1, 16); a2 += __shfl_xor(a2, 1, 16);
      a1 += __shfl_xor(a1, 2, 16); a2 += __shfl_xor(a2, 2, 16);
      a1 += __shfl_xor(a1, 4, 16); a2 += __shfl_xor(a2, 4, 16);
      a1 += __shfl_xor(a1, 8, 16); a2 += __shfl_xor(a2, 8, 16);
      if (l16 == 0) { atomicAdd(&s1[c0 + r], a1); atomicAdd(&s2[c0 + r], a2); }
    }
  }
  __syncthreads();
  if (tid < kCH) { atomicAdd(&gS1[tid], s1[tid]); atomicAdd(&gS2[tid], s2[tid]); }
}

// ---------------------------------------------------------------------------
// Kernel 3: BatchNorm finalize (in-place on d_out), float4 vectorized
// ---------------------------------------------------------------------------
__global__ __launch_bounds__(256) void k_bn(
    float* __restrict__ y, const float* __restrict__ gS1,
    const float* __restrict__ gS2, const float* __restrict__ gamma,
    const float* __restrict__ beta)
{
  const size_t i4   = (size_t)blockIdx.x * 256 + threadIdx.x;
  const size_t base = i4 * 4;
  const int c       = (int)((base / kN) % kCH);  // kN % 4 == 0 -> same c for all 4
  const float cnt   = (float)kB * (float)kN;
  const float mean  = gS1[c] / cnt;
  const float var   = gS2[c] / cnt - mean * mean;
  const float sc    = rsqrtf(var + 1e-5f) * gamma[c];
  const float sh    = beta[c] - mean * sc;
  float4 v = *(float4*)(y + base);
  v.x = v.x * sc + sh; v.y = v.y * sc + sh;
  v.z = v.z * sc + sh; v.w = v.w * sc + sh;
  *(float4*)(y + base) = v;
}

extern "C" void kernel_launch(void* const* d_in, const int* in_sizes, int n_in,
                              void* d_out, int out_size, void* d_ws, size_t ws_size,
                              hipStream_t stream) {
  const float* x     = (const float*)d_in[0];
  const float* Wq    = (const float*)d_in[1];
  const float* Wk    = (const float*)d_in[2];
  const float* Wv    = (const float*)d_in[3];
  const float* Wout  = (const float*)d_in[4];
  const float* gamma = (const float*)d_in[5];
  const float* beta  = (const float*)d_in[6];
  float* out = (float*)d_out;

  // workspace layout: [KV | Ksum | S1 | S2 | Q]
  float* wsf = (float*)d_ws;
  float* gKV = wsf;
  float* gKs = gKV + (size_t)kB * kMID * kMID;
  float* gS1 = gKs + (size_t)kB * kMID;
  float* gS2 = gS1 + kCH;
  float* gQ  = gS2 + kCH;
  const size_t zero_bytes =
      ((size_t)kB * kMID * kMID + (size_t)kB * kMID + 2 * kCH) * sizeof(float);
  hipMemsetAsync(d_ws, 0, zero_bytes, stream);

  dim3 grid(kN / kNT, kB);
  k_qkv<<<grid, 256, 0, stream>>>(x, Wq, Wk, Wv, gQ, gKV, gKs);
  k_attn<<<grid, 256, 0, stream>>>(gQ, gKV, gKs, Wout, out, gS1, gS2);

  const size_t total = (size_t)kB * kCH * kN;
  k_bn<<<(unsigned)(total / 4 / 256), 256, 0, stream>>>(out, gS1, gS2, gamma, beta);
}
